// OMGNN_RNN_74560632259077
// MI455X (gfx1250) — matrix operand order repaired and checked by reading, not compiled, not run on hardware
//
#include <hip/hip_runtime.h>
#include <cstdint>
#include <cstddef>

// ---------------------------------------------------------------------------
// OMGNN hierarchical bond-message-passing, CDNA5 (gfx1250, wave32, WMMA).
// All GEMMs run through v_wmma_f32_16x16x32_bf16 (bf16 A/B, f32 accumulate).
// GEMM wave tile: 32(M) x 64(N) -> 8 accumulators, 1.5 b128 loads per WMMA.
// ---------------------------------------------------------------------------

typedef __attribute__((ext_vector_type(16))) __bf16 v16bf;
typedef __attribute__((ext_vector_type(8)))  __bf16 v8bf;
typedef __attribute__((ext_vector_type(8)))  float  v8f;

#define CN1 102400
#define CE1 300000
#define CN2 10240
#define CE2 40960
#define CN3 1024
#define CE3 4096
#define CG  64
#define CD  256
#define CB  64

static inline int gridFor(long long n, int blk) { return (int)((n + blk - 1) / blk); }

// ---------------------------- utility kernels ------------------------------

__global__ void k_zero(float* __restrict__ p, int n) {
  int i = blockIdx.x * blockDim.x + threadIdx.x;
  if (i < n) p[i] = 0.0f;
}

// W [K x 256] f32 (row-major, fan_in x fan_out)  ->  Wt [256 x K] bf16
__global__ void k_wt(const float* __restrict__ W, __bf16* __restrict__ Wt, int K) {
  int i = blockIdx.x * blockDim.x + threadIdx.x;
  if (i >= K * 256) return;
  int k = i / 256;
  int n = i - k * 256;
  Wt[(size_t)n * K + k] = (__bf16)W[i];
}

// A0[e][0:256] = x[src[e]] ; A0[e][256:320] = ea[e]   (bf16)
__global__ void k_build_A0(const float* __restrict__ x, const float* __restrict__ ea,
                           const int* __restrict__ src, __bf16* __restrict__ A, int E) {
  int i = blockIdx.x * blockDim.x + threadIdx.x;
  if (i >= E * 320) return;
  int e = i / 320;
  int c = i - e * 320;
  float v = (c < 256) ? x[(size_t)src[e] * 256 + c]
                      : ea[(size_t)e * 64 + (c - 256)];
  A[i] = (__bf16)v;
}

// agg[dst[e]] += H[e]   (segment_sum over edges -> nodes)
__global__ void k_scatter_add(const float* __restrict__ H, const int* __restrict__ dst,
                              float* __restrict__ agg, int E) {
  int i = blockIdx.x * blockDim.x + threadIdx.x;
  if (i >= E * 256) return;
  int e = i >> 8;
  int d = i & 255;
  atomicAdd(&agg[(size_t)dst[e] * 256 + d], H[i]);
}

// Aw[e] = bf16( agg[src[e]] - H[rev[e]] )
__global__ void k_build_Aw(const float* __restrict__ agg, const float* __restrict__ H,
                           const int* __restrict__ src, const int* __restrict__ rev,
                           __bf16* __restrict__ A, int E) {
  int i = blockIdx.x * blockDim.x + threadIdx.x;
  if (i >= E * 256) return;
  int e = i >> 8;
  int d = i & 255;
  float v = agg[(size_t)src[e] * 256 + d] - H[(size_t)rev[e] * 256 + d];
  A[i] = (__bf16)v;
}

// Per node: s = sum(Mb[n]); Mn = (s==0) ? x[n] : Mb[n];
// A2[n][0:256] = x[n]; A2[n][256:512] = Mn   (bf16). One block (256 thr)/node.
__global__ void k_build_A2(const float* __restrict__ x, const float* __restrict__ Mb,
                           __bf16* __restrict__ A2) {
  __shared__ float red[256];
  int n = blockIdx.x;
  int d = threadIdx.x;
  float m = Mb[(size_t)n * 256 + d];
  red[d] = m;
  __syncthreads();
  #pragma unroll
  for (int s = 128; s > 0; s >>= 1) {
    if (d < s) red[d] += red[d + s];
    __syncthreads();
  }
  float rs = red[0];
  float xv = x[(size_t)n * 256 + d];
  float mv = (rs == 0.0f) ? xv : m;
  A2[(size_t)n * 512 + d]       = (__bf16)xv;
  A2[(size_t)n * 512 + 256 + d] = (__bf16)mv;
}

// mean-pool scatter + divide
__global__ void k_pool_scat(const float* __restrict__ h, const int* __restrict__ seg,
                            float* __restrict__ pooled, float* __restrict__ cnt, int Nin) {
  int i = blockIdx.x * blockDim.x + threadIdx.x;
  if (i >= Nin * 256) return;
  int n = i >> 8;
  int d = i & 255;
  int s = seg[n];
  atomicAdd(&pooled[(size_t)s * 256 + d], h[i]);
  if (d == 0) atomicAdd(&cnt[s], 1.0f);
}

__global__ void k_pool_div(float* __restrict__ pooled, const float* __restrict__ cnt, int Nout) {
  int i = blockIdx.x * blockDim.x + threadIdx.x;
  if (i >= Nout * 256) return;
  pooled[i] /= fmaxf(cnt[i >> 8], 1.0f);
}

// ------------------------------- WMMA GEMM ---------------------------------
// C[M x 256] = relu( A[M x K](bf16) @ W[K x 256] + bias (+ skip) )
// Wt is W transposed: [256 x K] bf16 (contiguous along K).
// Block = 128 threads (4 waves). Block tile: 32 (M) x 256 (N); wave: 32 x 64.

__device__ __forceinline__ v16bf load_frag_a(const __bf16* p) {
  // A 16x32 bf16 layout: elems 0..15 -> K = hi*16 + 0..15 (contiguous)
  union { v16bf f; v8bf h[2]; } u;
  u.h[0] = *reinterpret_cast<const v8bf*>(p);
  u.h[1] = *reinterpret_cast<const v8bf*>(p + 8);
  return u.f;
}

__device__ __forceinline__ v16bf load_frag_b(const __bf16* p) {
  // B 32x16 bf16 layout: elems 0..15 -> K = hi*16 + 0..15 (contiguous)
  union { v16bf f; v8bf h[2]; } u;
  u.h[0] = *reinterpret_cast<const v8bf*>(p);
  u.h[1] = *reinterpret_cast<const v8bf*>(p + 8);
  return u.f;
}

__global__ void __launch_bounds__(128)
k_gemm_bias_relu(const __bf16* __restrict__ A, const __bf16* __restrict__ Wt,
                 const float* __restrict__ bias, const float* __restrict__ skip,
                 float* __restrict__ C, int K) {
  const int lane = threadIdx.x & 31;
  const int wave = threadIdx.x >> 5;
  const int l15  = lane & 15;
  const int hi   = lane >> 4;
  const int m0   = blockIdx.x * 32;
  const int n0   = wave * 64;

  const __bf16* Arow0 = A  + (size_t)(m0 +      l15) * K + hi * 16;
  const __bf16* Arow1 = A  + (size_t)(m0 + 16 + l15) * K + hi * 16;
  const __bf16* B0    = Wt + (size_t)(n0 +  0 + l15) * K + hi * 16;
  const __bf16* B1    = Wt + (size_t)(n0 + 16 + l15) * K + hi * 16;
  const __bf16* B2    = Wt + (size_t)(n0 + 32 + l15) * K + hi * 16;
  const __bf16* B3    = Wt + (size_t)(n0 + 48 + l15) * K + hi * 16;

  v8f acc[2][4];
  #pragma unroll
  for (int g = 0; g < 2; ++g)
    #pragma unroll
    for (int j = 0; j < 4; ++j)
      acc[g][j] = (v8f){};

  for (int k0 = 0; k0 < K; k0 += 32) {
    v16bf a0 = load_frag_a(Arow0 + k0);
    v16bf a1 = load_frag_a(Arow1 + k0);
    v16bf b0 = load_frag_b(B0 + k0);
    v16bf b1 = load_frag_b(B1 + k0);
    v16bf b2 = load_frag_b(B2 + k0);
    v16bf b3 = load_frag_b(B3 + k0);
    acc[0][0] = __builtin_amdgcn_wmma_f32_16x16x32_bf16(false, a0, false, b0, (short)0, acc[0][0], false, false);
    acc[0][1] = __builtin_amdgcn_wmma_f32_16x16x32_bf16(false, a0, false, b1, (short)0, acc[0][1], false, false);
    acc[0][2] = __builtin_amdgcn_wmma_f32_16x16x32_bf16(false, a0, false, b2, (short)0, acc[0][2], false, false);
    acc[0][3] = __builtin_amdgcn_wmma_f32_16x16x32_bf16(false, a0, false, b3, (short)0, acc[0][3], false, false);
    acc[1][0] = __builtin_amdgcn_wmma_f32_16x16x32_bf16(false, a1, false, b0, (short)0, acc[1][0], false, false);
    acc[1][1] = __builtin_amdgcn_wmma_f32_16x16x32_bf16(false, a1, false, b1, (short)0, acc[1][1], false, false);
    acc[1][2] = __builtin_amdgcn_wmma_f32_16x16x32_bf16(false, a1, false, b2, (short)0, acc[1][2], false, false);
    acc[1][3] = __builtin_amdgcn_wmma_f32_16x16x32_bf16(false, a1, false, b3, (short)0, acc[1][3], false, false);
  }

  const bool has_skip = (skip != nullptr);
  #pragma unroll
  for (int g = 0; g < 2; ++g) {
    #pragma unroll
    for (int r = 0; r < 8; ++r) {
      const int row = m0 + g * 16 + hi * 8 + r;
      float* Crow = C + (size_t)row * 256;
      const float* Srow = has_skip ? (skip + (size_t)row * 256) : nullptr;
      const int c0 = n0 + l15;
      float v;
      v = acc[g][0][r] + bias[c0];      if (has_skip) v += Srow[c0];      Crow[c0]      = v > 0.f ? v : 0.f;
      v = acc[g][1][r] + bias[c0 + 16]; if (has_skip) v += Srow[c0 + 16]; Crow[c0 + 16] = v > 0.f ? v : 0.f;
      v = acc[g][2][r] + bias[c0 + 32]; if (has_skip) v += Srow[c0 + 32]; Crow[c0 + 32] = v > 0.f ? v : 0.f;
      v = acc[g][3][r] + bias[c0 + 48]; if (has_skip) v += Srow[c0 + 48]; Crow[c0 + 48] = v > 0.f ? v : 0.f;
    }
  }
}

// ----------------------------- level driver --------------------------------

static void run_level(const float* x, const float* ea,
                      const int* ei, const int* rev,
                      int Nn, int Ne, int depth,
                      const float* Wi, const float* bi,
                      const float* Wh, const float* bh,
                      const float* Wo, const float* bo,
                      __bf16* WiT, __bf16* WhT, __bf16* WoT,
                      __bf16* bufA, float* H0, float* H, float* agg,
                      hipStream_t stream) {
  const int* src = ei;
  const int* dst = ei + Ne;

  k_wt<<<gridFor(320 * 256, 256), 256, 0, stream>>>(Wi, WiT, 320);
  k_wt<<<gridFor(256 * 256, 256), 256, 0, stream>>>(Wh, WhT, 256);
  k_wt<<<gridFor(512 * 256, 256), 256, 0, stream>>>(Wo, WoT, 512);

  // H0 = relu(concat(x[src], ea) @ Wi + bi)
  k_build_A0<<<gridFor((long long)Ne * 320, 256), 256, 0, stream>>>(x, ea, src, bufA, Ne);
  k_gemm_bias_relu<<<dim3(Ne / 32), 128, 0, stream>>>(bufA, WiT, bi, nullptr, H0, 320);

  const float* Hc = H0;
  for (int it = 1; it < depth; ++it) {
    k_zero<<<gridFor((long long)Nn * 256, 256), 256, 0, stream>>>(agg, Nn * 256);
    k_scatter_add<<<gridFor((long long)Ne * 256, 256), 256, 0, stream>>>(Hc, dst, agg, Ne);
    k_build_Aw<<<gridFor((long long)Ne * 256, 256), 256, 0, stream>>>(agg, Hc, src, rev, bufA, Ne);
    // H = relu(H0 + M @ Wh + bh)   (in-place safe: GEMM reads only bufA/H0)
    k_gemm_bias_relu<<<dim3(Ne / 32), 128, 0, stream>>>(bufA, WhT, bh, H0, H, 256);
    Hc = H;
  }

  // M = segment_sum(H, dst); where(rowsum==0, x, M); out = relu(concat(x,M)@Wo+bo)
  k_zero<<<gridFor((long long)Nn * 256, 256), 256, 0, stream>>>(agg, Nn * 256);
  k_scatter_add<<<gridFor((long long)Ne * 256, 256), 256, 0, stream>>>(Hc, dst, agg, Ne);
  k_build_A2<<<dim3(Nn), 256, 0, stream>>>(x, agg, bufA);
  k_gemm_bias_relu<<<dim3(Nn / 32), 128, 0, stream>>>(bufA, WoT, bo, nullptr, H0, 512);
  // node outputs now live in H0 ([Nn x 256])
}

static void mean_pool(const float* h, const int* seg, float* pooled, float* cnt,
                      int Nin, int Nout, hipStream_t stream) {
  k_zero<<<gridFor((long long)Nout * 256, 256), 256, 0, stream>>>(pooled, Nout * 256);
  k_zero<<<gridFor(Nout, 256), 256, 0, stream>>>(cnt, Nout);
  k_pool_scat<<<gridFor((long long)Nin * 256, 256), 256, 0, stream>>>(h, seg, pooled, cnt, Nin);
  k_pool_div<<<gridFor((long long)Nout * 256, 256), 256, 0, stream>>>(pooled, cnt, Nout);
}

// ------------------------------- entry point -------------------------------

extern "C" void kernel_launch(void* const* d_in, const int* in_sizes, int n_in,
                              void* d_out, int out_size, void* d_ws, size_t ws_size,
                              hipStream_t stream) {
  (void)in_sizes; (void)n_in; (void)out_size; (void)ws_size;

  const float* x   = (const float*)d_in[0];
  const float* ea1 = (const float*)d_in[1];
  const float* ea2 = (const float*)d_in[2];
  const float* ea3 = (const float*)d_in[3];
  const int* ei1   = (const int*)d_in[4];
  const int* rev1  = (const int*)d_in[5];
  const int* ei2   = (const int*)d_in[6];
  const int* rev2  = (const int*)d_in[7];
  const int* ei3   = (const int*)d_in[8];
  const int* rev3  = (const int*)d_in[9];
  const int* b12   = (const int*)d_in[10];
  const int* b2    = (const int*)d_in[11];
  const int* b3    = (const int*)d_in[12];

  const float* Wp[18];
  for (int i = 0; i < 18; ++i) Wp[i] = (const float*)d_in[13 + i];
  // per level: Wi, bi, Wh, bh, Wo, bo

  // workspace carve-up
  size_t off = 0;
  auto alloc = [&](size_t bytes) -> char* {
    char* p = (char*)d_ws + off;
    off = (off + bytes + 255) & ~(size_t)255;
    return p;
  };
  __bf16* WiT[3]; __bf16* WhT[3]; __bf16* WoT[3];
  for (int l = 0; l < 3; ++l) {
    WiT[l] = (__bf16*)alloc((size_t)320 * 256 * 2);
    WhT[l] = (__bf16*)alloc((size_t)256 * 256 * 2);
    WoT[l] = (__bf16*)alloc((size_t)512 * 256 * 2);
  }
  float* pooled2 = (float*)alloc((size_t)CN2 * 256 * 4);
  float* cnt2    = (float*)alloc((size_t)CN2 * 4);
  float* pooled3 = (float*)alloc((size_t)CN3 * 256 * 4);
  float* cnt3    = (float*)alloc((size_t)CN3 * 4);
  float* cntG    = (float*)alloc((size_t)CG * 4);
  __bf16* bufA   = (__bf16*)alloc((size_t)CE1 * 320 * 2);   // shared A operand buffer
  float* H0      = (float*)alloc((size_t)CE1 * 256 * 4);    // H0 / node-output reuse
  float* H       = (float*)alloc((size_t)CE1 * 256 * 4);
  float* agg     = (float*)alloc((size_t)CN1 * 256 * 4);    // segment sums / M buffer

  // Level 1
  run_level(x, ea1, ei1, rev1, CN1, CE1, 3,
            Wp[0], Wp[1], Wp[2], Wp[3], Wp[4], Wp[5],
            WiT[0], WhT[0], WoT[0], bufA, H0, H, agg, stream);
  mean_pool(H0, b12, pooled2, cnt2, CN1, CN2, stream);

  // Level 2
  run_level(pooled2, ea2, ei2, rev2, CN2, CE2, 3,
            Wp[6], Wp[7], Wp[8], Wp[9], Wp[10], Wp[11],
            WiT[1], WhT[1], WoT[1], bufA, H0, H, agg, stream);
  mean_pool(H0, b2, pooled3, cnt3, CN2, CN3, stream);

  // Level 3
  run_level(pooled3, ea3, ei3, rev3, CN3, CE3, 4,
            Wp[12], Wp[13], Wp[14], Wp[15], Wp[16], Wp[17],
            WiT[2], WhT[2], WoT[2], bufA, H0, H, agg, stream);
  mean_pool(H0, b3, (float*)d_out, cntG, CN3, CG, stream);
}
